// MixtralMoE_87686052315716
// MI455X (gfx1250) — compile-verified, matrix operand order checked by reading, not run on hardware
//
#include <hip/hip_runtime.h>

#define TOKENS 16384
#define HIDDEN 1024
#define INTER  3584
#define NEXP   8
#define KC     64   // K-chunk staged in LDS per iteration

typedef __attribute__((ext_vector_type(16))) __bf16 v16bf;
typedef __attribute__((ext_vector_type(8)))  float  v8f;

// ---------------------------------------------------------------- helpers
__device__ __forceinline__ unsigned short f2bf(float f) {
  // round-to-nearest-even f32 -> bf16
  unsigned int b = __builtin_bit_cast(unsigned int, f);
  b += 0x7FFFu + ((b >> 16) & 1u);
  return (unsigned short)(b >> 16);
}

// Load a 16x32 bf16 fragment (A-layout; B uses the symmetric N-per-lane layout)
// from LDS. p0 points at (row0, k0); stride in halves. Per ISA 7.12.2:
// lanes 0-15 hold rows 0-15 with K 0..7 & 16..23; lanes 16-31 hold K 8..15 & 24..31.
__device__ __forceinline__ v16bf load_frag(const unsigned short* p0, int stride) {
  int lane = threadIdx.x & 31;
  const unsigned short* p = p0 + (lane & 15) * stride + ((lane >> 4) << 3);
  union { uint4 q[2]; v16bf f; } u;
  u.q[0] = *(const uint4*)(p);       // halves k .. k+7
  u.q[1] = *(const uint4*)(p + 16);  // halves k+16 .. k+23
  return u.f;
}

__device__ __forceinline__ v8f wmma_bf16(v16bf a, v16bf b, v8f c) {
  return __builtin_amdgcn_wmma_f32_16x16x32_bf16(false, a, false, b, (short)0, c,
                                                 false, false);
}

// ---------------------------------------------------------------- zero out + counters
__global__ void __launch_bounds__(256) zero_kernel(float* out, int* counts, size_t n) {
  size_t i = (size_t)blockIdx.x * blockDim.x + threadIdx.x;
  size_t stride = (size_t)gridDim.x * blockDim.x;
  for (; i < n; i += stride) out[i] = 0.0f;
  if (blockIdx.x == 0 && threadIdx.x < NEXP) counts[threadIdx.x] = 0;
}

// ---------------------------------------------------------------- f32 -> bf16 convert
__global__ void __launch_bounds__(256) cvt_kernel(const float* __restrict__ src,
                                                  unsigned short* __restrict__ dst,
                                                  size_t n4) {
  size_t i = (size_t)blockIdx.x * blockDim.x + threadIdx.x;
  size_t stride = (size_t)gridDim.x * blockDim.x;
  for (; i < n4; i += stride) {
    float4 f = ((const float4*)src)[i];
    uint2 o;
    o.x = (unsigned)f2bf(f.x) | ((unsigned)f2bf(f.y) << 16);
    o.y = (unsigned)f2bf(f.z) | ((unsigned)f2bf(f.w) << 16);
    ((uint2*)dst)[i] = o;
  }
}

// ---------------------------------------------------------------- router: wave per token
__global__ void __launch_bounds__(256) router_kernel(const float* __restrict__ x,
                                                     const float* __restrict__ gw,
                                                     float* __restrict__ combineW,
                                                     int* __restrict__ counts,
                                                     int* __restrict__ lists) {
  int wave = threadIdx.x >> 5;
  int lane = threadIdx.x & 31;
  int t = blockIdx.x * 8 + wave;
  if (t >= TOKENS) return;

  float acc[NEXP] = {0.f, 0.f, 0.f, 0.f, 0.f, 0.f, 0.f, 0.f};
  for (int h = lane; h < HIDDEN; h += 32) {
    float xv = x[(size_t)t * HIDDEN + h];
#pragma unroll
    for (int e = 0; e < NEXP; ++e) acc[e] += xv * gw[e * HIDDEN + h];
  }
#pragma unroll
  for (int e = 0; e < NEXP; ++e)
    for (int off = 16; off; off >>= 1) acc[e] += __shfl_xor(acc[e], off, 32);

  if (lane == 0) {
    float m = acc[0];
#pragma unroll
    for (int e = 1; e < NEXP; ++e) m = fmaxf(m, acc[e]);
    float p[NEXP];
#pragma unroll
    for (int e = 0; e < NEXP; ++e) p[e] = __expf(acc[e] - m);
    int i0 = 0;
#pragma unroll
    for (int e = 1; e < NEXP; ++e) if (p[e] > p[i0]) i0 = e;
    int i1 = (i0 == 0) ? 1 : 0;
#pragma unroll
    for (int e = 0; e < NEXP; ++e) if (e != i0 && p[e] > p[i1]) i1 = e;
    // top-2 softmax weights renormalized == exp ratio (denominators cancel)
    float inv = 1.0f / (p[i0] + p[i1]);
    combineW[t * 2 + 0] = p[i0] * inv;
    combineW[t * 2 + 1] = p[i1] * inv;
    int pos0 = atomicAdd(&counts[i0], 1);
    lists[i0 * TOKENS + pos0] = t * 2 + 0;
    int pos1 = atomicAdd(&counts[i1], 1);
    lists[i1 * TOKENS + pos1] = t * 2 + 1;
  }
}

// ---------------------------------------------------------------- pass A: gate+up+SwiGLU
// Block tile: 64 gathered token-slots x 128 inter cols, K = HIDDEN.
// 8 waves laid out 4(M) x 2(N); each wave: 16 rows x 64 cols -> 4 WMMA subtiles x2 mats.
__global__ void __launch_bounds__(256) moe_gateup(const unsigned short* __restrict__ Xb,
                                                  const unsigned short* __restrict__ W1b,
                                                  const unsigned short* __restrict__ W3b,
                                                  const int* __restrict__ counts,
                                                  const int* __restrict__ lists,
                                                  unsigned short* __restrict__ act) {
  int e = blockIdx.z;
  int cnt = counts[e];
  int m0 = blockIdx.x * 64;
  if (m0 >= cnt) return;
  int n0 = blockIdx.y * 128;

  __shared__ __align__(16) unsigned short aT[64 * KC];
  __shared__ __align__(16) unsigned short b1T[128 * KC];
  __shared__ __align__(16) unsigned short b3T[128 * KC];
  __shared__ int slotS[64];
  __shared__ int tokS[64];

  int tid = threadIdx.x;
  if (tid < 64) {
    int r = m0 + tid;
    int s = (r < cnt) ? lists[e * TOKENS + r] : 0;  // clamp invalid rows to slot 0
    slotS[tid] = s;
    tokS[tid] = s >> 1;
  }
  __syncthreads();

  int wave = tid >> 5;
  int wm = (wave & 3) * 16;   // wave row offset within 64
  int wn = (wave >> 2) * 64;  // wave col offset within 128

  const v8f vz = {0.f, 0.f, 0.f, 0.f, 0.f, 0.f, 0.f, 0.f};
  v8f acc1[4] = {vz, vz, vz, vz};
  v8f acc3[4] = {vz, vz, vz, vz};

  for (int kc = 0; kc < HIDDEN; kc += KC) {
    __syncthreads();  // previous tile fully consumed
#pragma unroll
    for (int j = 0; j < 2; ++j) {  // A: 64x64 halves = 512 uint4
      int idx = tid + j * 256;
      int r = idx >> 3;
      int kk = (idx & 7) * 8;
      *(uint4*)&aT[r * KC + kk] =
          *(const uint4*)&Xb[(size_t)tokS[r] * HIDDEN + kc + kk];
    }
#pragma unroll
    for (int j = 0; j < 4; ++j) {  // B1/B3: 128x64 halves = 1024 uint4 each
      int idx = tid + j * 256;
      int r = idx >> 3;
      int kk = (idx & 7) * 8;
      const unsigned short* g1 = &W1b[((size_t)e * INTER + n0 + r) * HIDDEN + kc + kk];
      const unsigned short* g3 = &W3b[((size_t)e * INTER + n0 + r) * HIDDEN + kc + kk];
      *(uint4*)&b1T[r * KC + kk] = *(const uint4*)g1;
      *(uint4*)&b3T[r * KC + kk] = *(const uint4*)g3;
      if (kc + KC < HIDDEN) {  // prefetch next K panel -> global_prefetch_b8
        __builtin_prefetch(g1 + KC, 0, 0);
        __builtin_prefetch(g3 + KC, 0, 0);
      }
    }
    __syncthreads();
#pragma unroll
    for (int ks = 0; ks < KC; ks += 32) {
      // Batch all fragment loads for this k-step so the ds_load_b128s issue
      // back-to-back and the WMMAs overlap with partial dscnt waits.
      v16bf a = load_frag(&aT[wm * KC + ks], KC);
      v16bf b1[4], b3[4];
#pragma unroll
      for (int s = 0; s < 4; ++s) {
        b1[s] = load_frag(&b1T[(wn + s * 16) * KC + ks], KC);
        b3[s] = load_frag(&b3T[(wn + s * 16) * KC + ks], KC);
      }
#pragma unroll
      for (int s = 0; s < 4; ++s) {
        acc1[s] = wmma_bf16(a, b1[s], acc1[s]);
        acc3[s] = wmma_bf16(a, b3[s], acc3[s]);
      }
    }
  }

  // SwiGLU epilogue + bf16 store, act indexed by slot id (t*2+k)
  int lane = tid & 31;
  int rsel = (lane >> 4) * 8;
  int nlane = lane & 15;
#pragma unroll
  for (int s = 0; s < 4; ++s) {
#pragma unroll
    for (int i = 0; i < 8; ++i) {
      int mrow = wm + i + rsel;  // 0..63
      if (m0 + mrow < cnt) {
        float h = acc1[s][i];
        float u = acc3[s][i];
        float v = (h / (1.0f + __expf(-h))) * u;
        int col = n0 + wn + s * 16 + nlane;
        act[(size_t)slotS[mrow] * INTER + col] = f2bf(v);
      }
    }
  }
}

// ---------------------------------------------------------------- pass B: down proj + scatter
__global__ void __launch_bounds__(256) moe_down(const unsigned short* __restrict__ act,
                                                const unsigned short* __restrict__ W2b,
                                                const int* __restrict__ counts,
                                                const int* __restrict__ lists,
                                                const float* __restrict__ combineW,
                                                float* __restrict__ out) {
  int e = blockIdx.z;
  int cnt = counts[e];
  int m0 = blockIdx.x * 64;
  if (m0 >= cnt) return;
  int n0 = blockIdx.y * 128;  // hidden column tile

  __shared__ __align__(16) unsigned short aT[64 * KC];
  __shared__ __align__(16) unsigned short bT[128 * KC];
  __shared__ int slotS[64];

  int tid = threadIdx.x;
  if (tid < 64) {
    int r = m0 + tid;
    slotS[tid] = (r < cnt) ? lists[e * TOKENS + r] : 0;
  }
  __syncthreads();

  int wave = tid >> 5;
  int wm = (wave & 3) * 16;
  int wn = (wave >> 2) * 64;

  const v8f vz = {0.f, 0.f, 0.f, 0.f, 0.f, 0.f, 0.f, 0.f};
  v8f acc[4] = {vz, vz, vz, vz};

  for (int kc = 0; kc < INTER; kc += KC) {
    __syncthreads();
#pragma unroll
    for (int j = 0; j < 2; ++j) {
      int idx = tid + j * 256;
      int r = idx >> 3;
      int kk = (idx & 7) * 8;
      *(uint4*)&aT[r * KC + kk] =
          *(const uint4*)&act[(size_t)slotS[r] * INTER + kc + kk];
    }
#pragma unroll
    for (int j = 0; j < 4; ++j) {
      int idx = tid + j * 256;
      int r = idx >> 3;
      int kk = (idx & 7) * 8;
      const unsigned short* g = &W2b[((size_t)e * HIDDEN + n0 + r) * INTER + kc + kk];
      *(uint4*)&bT[r * KC + kk] = *(const uint4*)g;
      if (kc + KC < INTER) __builtin_prefetch(g + KC, 0, 0);
    }
    __syncthreads();
#pragma unroll
    for (int ks = 0; ks < KC; ks += 32) {
      v16bf a = load_frag(&aT[wm * KC + ks], KC);
      v16bf b[4];
#pragma unroll
      for (int s = 0; s < 4; ++s)
        b[s] = load_frag(&bT[(wn + s * 16) * KC + ks], KC);
#pragma unroll
      for (int s = 0; s < 4; ++s)
        acc[s] = wmma_bf16(a, b[s], acc[s]);
    }
  }

  int lane = tid & 31;
  int rsel = (lane >> 4) * 8;
  int nlane = lane & 15;
#pragma unroll
  for (int s = 0; s < 4; ++s) {
#pragma unroll
    for (int i = 0; i < 8; ++i) {
      int mrow = wm + i + rsel;
      if (m0 + mrow < cnt) {
        int slot = slotS[mrow];
        float cw = combineW[slot];
        int t = slot >> 1;
        int col = n0 + wn + s * 16 + nlane;
        __hip_atomic_fetch_add(&out[(size_t)t * HIDDEN + col], acc[s][i] * cw,
                               __ATOMIC_RELAXED, __HIP_MEMORY_SCOPE_AGENT);
      }
    }
  }
}

// ---------------------------------------------------------------- launch
extern "C" void kernel_launch(void* const* d_in, const int* in_sizes, int n_in,
                              void* d_out, int out_size, void* d_ws, size_t ws_size,
                              hipStream_t stream) {
  const float* x  = (const float*)d_in[0];
  const float* gw = (const float*)d_in[1];
  const float* w1 = (const float*)d_in[2];
  const float* w3 = (const float*)d_in[3];
  const float* w2 = (const float*)d_in[4];
  float* out = (float*)d_out;

  // Workspace layout (~425 MB total)
  char* ws = (char*)d_ws;
  size_t off = 0;
  int* counts = (int*)(ws + off);            off += 256;
  int* lists  = (int*)(ws + off);            off += (size_t)NEXP * TOKENS * 4;
  float* combineW = (float*)(ws + off);      off += (size_t)TOKENS * 2 * 4;
  unsigned short* Xb  = (unsigned short*)(ws + off); off += (size_t)TOKENS * HIDDEN * 2;
  unsigned short* W1b = (unsigned short*)(ws + off); off += (size_t)NEXP * INTER * HIDDEN * 2;
  unsigned short* W3b = (unsigned short*)(ws + off); off += (size_t)NEXP * INTER * HIDDEN * 2;
  unsigned short* W2b = (unsigned short*)(ws + off); off += (size_t)NEXP * HIDDEN * INTER * 2;
  unsigned short* act = (unsigned short*)(ws + off); off += (size_t)TOKENS * 2 * INTER * 2;
  (void)in_sizes; (void)n_in; (void)out_size; (void)ws_size;

  zero_kernel<<<2048, 256, 0, stream>>>(out, counts, (size_t)TOKENS * HIDDEN);
  cvt_kernel<<<2048, 256, 0, stream>>>(x,  Xb,  (size_t)TOKENS * HIDDEN / 4);
  cvt_kernel<<<4096, 256, 0, stream>>>(w1, W1b, (size_t)NEXP * INTER * HIDDEN / 4);
  cvt_kernel<<<4096, 256, 0, stream>>>(w3, W3b, (size_t)NEXP * INTER * HIDDEN / 4);
  cvt_kernel<<<4096, 256, 0, stream>>>(w2, W2b, (size_t)NEXP * HIDDEN * INTER / 4);
  router_kernel<<<TOKENS / 8, 256, 0, stream>>>(x, gw, combineW, counts, lists);

  dim3 gA(TOKENS / 64, INTER / 128, NEXP);
  moe_gateup<<<gA, 256, 0, stream>>>(Xb, W1b, W3b, counts, lists, act);

  dim3 gB(TOKENS / 64, HIDDEN / 128, NEXP);
  moe_down<<<gB, 256, 0, stream>>>(act, W2b, counts, lists, combineW, out);
}
